// FRNNPathB_55259049230415
// MI455X (gfx1250) — matrix operand order repaired and checked by reading, not compiled, not run on hardware
//
#include <hip/hip_runtime.h>
#include <stdint.h>

// ---------------- problem constants ----------------
static constexpr int Bsz  = 8;
static constexpr int Ssz  = 2048;
static constexpr int DIN  = 1024;
static constexpr int Hdim = 2048;
static constexpr int Kmod = 256;    // K modes
static constexpr int DM   = 512;
static constexpr int DOUT = 1024;
static constexpr int BANK = 4096;
static constexpr int MTOT = Bsz * Ssz;   // 16384 rows
static constexpr float STICKINESS = 0.1f;
static constexpr float BANK_SCALE = 4.0f;
static constexpr float RMS_EPS    = 1e-6f;
static constexpr float NEG_BIG    = -3.402823466e38f;

// ---------------- WMMA types ----------------
typedef __attribute__((ext_vector_type(8)))  float  v8f;
typedef __attribute__((ext_vector_type(16))) __bf16 v16bf;
typedef __attribute__((ext_vector_type(4)))  int    si4;
typedef __attribute__((ext_vector_type(8)))  int    si8;

union FragAB {
    uint4  q[2];   // two 16B chunks (8 bf16 each)
    v16bf  v;
};

static __device__ __forceinline__ unsigned short f32_to_bf16(float f) {
    unsigned int u = __float_as_uint(f);
    unsigned int lsb = (u >> 16) & 1u;
    u += 0x7fffu + lsb;              // round-to-nearest-even
    return (unsigned short)(u >> 16);
}

// ---- CDNA5 async global->LDS copy (ASYNCcnt-tracked, no VGPR transit) ----
static __device__ __forceinline__ void async_copy_b128(unsigned int lds_addr,
                                                       const void* gaddr) {
    asm volatile("global_load_async_to_lds_b128 %0, %1, off"
                 :: "v"(lds_addr), "v"(gaddr) : "memory");
}
static __device__ __forceinline__ void wait_async0() {
    asm volatile("s_wait_asynccnt 0x0" ::: "memory");
}

// ---- CDNA5 Tensor Data Mover: 2-D tile load, bf16, padded LDS rows -------
// D# per ISA 8.3/8.4: group0 = {count/type/lds/global}, group1 = dims/strides.
// Padding: after every 32 DWORDs (128 B data row) insert 4 DWORDs (16 B)
// -> matches LROW = 144 B padded LDS row stride used by the WMMA frag reads.
static __device__ __forceinline__ void tdm_load_2d_bf16(
        unsigned int lds_off,        // LDS byte address of tile start
        const void* gptr,            // global byte address of tile start
        unsigned int tile_x,         // tile dim0 (elements, contiguous)
        unsigned int tile_y,         // tile dim1 (rows)
        unsigned int tensor_x,       // tensor dim0 (elements)
        unsigned int tensor_y,       // tensor dim1 (rows)
        unsigned int row_stride) {   // tensor dim0 stride (elements)
    const unsigned long long ga = (unsigned long long)(uintptr_t)gptr;
    si4 g0;
    g0[0] = 1;                                        // count=1, user mode
    g0[1] = (int)lds_off;                             // lds_addr
    g0[2] = (int)(unsigned int)(ga & 0xffffffffu);    // global_addr[31:0]
    g0[3] = (int)(((unsigned int)(ga >> 32) & 0x01ffffffu) | (2u << 30)); // type=2
    si8 g1;
    g1[0] = (int)((1u << 16)      // data_size = 2 bytes (bf16)
                | (1u << 20)      // pad_enable
                | (4u << 22)      // pad_interval: 32 DWORDs (128 B)
                | (3u << 25));    // pad_amount:   4 DWORDs (16 B)
    g1[1] = (int)((tensor_x & 0xffffu) << 16);                     // dim0 lo16
    g1[2] = (int)((tensor_x >> 16) | ((tensor_y & 0xffffu) << 16));
    g1[3] = (int)((tensor_y >> 16) | (tile_x << 16));              // tile_dim0
    g1[4] = (int)(tile_y);                                         // tile_dim1 (dim2=0)
    g1[5] = (int)row_stride;                                       // dim0_stride lo32
    g1[6] = 0;                                                     // stride hi / dim1_stride
    g1[7] = 0;
    asm volatile("tensor_load_to_lds %0, %1" :: "s"(g0), "s"(g1) : "memory");
}

// ---------------- conversion kernels ----------------
__global__ void k_f32_to_bf16(const float* __restrict__ src,
                              unsigned short* __restrict__ dst, int n) {
    int i = blockIdx.x * blockDim.x + threadIdx.x;
    int stride = gridDim.x * blockDim.x;
    for (; i < n; i += stride) dst[i] = f32_to_bf16(src[i]);
}

// src[rows][cols] (f32) -> dst[cols][rows] (bf16)
__global__ void k_transpose_bf16(const float* __restrict__ src,
                                 unsigned short* __restrict__ dst,
                                 int rows, int cols) {
    int i = blockIdx.x * blockDim.x + threadIdx.x;
    int total = rows * cols;
    int stride = gridDim.x * blockDim.x;
    for (; i < total; i += stride) {
        int c = i / rows;
        int r = i - c * rows;
        dst[i] = f32_to_bf16(src[(size_t)r * cols + c]);
    }
}

// ---------------- bf16 WMMA GEMM (LDS double-buffered) --------------------
// A tile staged by the Tensor Data Mover (TENSORcnt), B tile by per-lane
// async global->LDS copies (ASYNCcnt). Block tile 64(M) x 128(N), TK=64.
// 8 waves as 2(M) x 4(N); each wave computes 32x32 (2x2 WMMA accumulators).
static constexpr int TM = 64;
static constexpr int TN = 128;
static constexpr int TK = 64;
static constexpr int LROW = TK * 2 + 16;              // 144 B padded row stride
static constexpr int A_BYTES = TM * LROW;             // 9216
static constexpr int B_BYTES = TN * LROW;             // 18432
static constexpr int STAGE_BYTES = A_BYTES + B_BYTES; // 27648 (x2 = 54 KB)

static __device__ __forceinline__ void issue_stage_b(
        const unsigned short* __restrict__ Bt,
        int Kd, int nBlk, int kb, unsigned char* sB, int tid) {
    // B tile: 128 rows x 128 B = 1024 x 16B chunks -> 4 per thread
#pragma unroll
    for (int c = 0; c < 4; ++c) {
        const int chunk = tid + c * 256;
        const int row = chunk >> 3;
        const int ke  = (chunk & 7) * 8;
        const void* g = Bt + (size_t)(nBlk + row) * Kd + kb + ke;
        async_copy_b128((unsigned int)(uintptr_t)(sB + row * LROW + ke * 2), g);
    }
}

template<bool RELU, bool OUT_BF16>
__global__ __launch_bounds__(256)
void k_gemm_bf16(const unsigned short* __restrict__ A,
                 const unsigned short* __restrict__ Bt,
                 const float* __restrict__ bias,
                 void* __restrict__ Cout,
                 int Mrows, int Ncols, int Kd) {
    __shared__ unsigned char smem[2 * STAGE_BYTES];

    const int tid  = threadIdx.x;
    const int lane = tid & 31;
    const int wave = tid >> 5;
    const int wm   = wave & 1;          // 0..1  (M)
    const int wn   = wave >> 1;         // 0..3  (N)
    const int lg   = lane >> 4;         // half-wave group
    const int ln   = lane & 15;

    const int mBlk = blockIdx.x * TM;
    const int nBlk = blockIdx.y * TN;

    v8f acc[2][2];
#pragma unroll
    for (int mi = 0; mi < 2; ++mi)
#pragma unroll
        for (int ni = 0; ni < 2; ++ni)
#pragma unroll
            for (int j = 0; j < 8; ++j) acc[mi][ni][j] = 0.0f;

    const int nstages = Kd / TK;

    // prologue: stage 0
    if (wave == 0)
        tdm_load_2d_bf16((unsigned int)(uintptr_t)smem,
                         A + (size_t)mBlk * Kd, TK, TM,
                         (unsigned int)Kd, (unsigned int)Mrows,
                         (unsigned int)Kd);
    issue_stage_b(Bt, Kd, nBlk, 0, smem + A_BYTES, tid);

    for (int s = 0; s < nstages; ++s) {
        wait_async0();                              // own B-tile copies done
        __builtin_amdgcn_s_wait_tensorcnt(0);       // wave0: A-tile DMA done
        __syncthreads();                            // whole stage resident

        if (s + 1 < nstages) {
            unsigned char* nbuf = smem + ((s + 1) & 1) * STAGE_BYTES;
            // safe: buffer (s+1)&1 last read at stage s-1, sealed by the
            // end-of-stage barrier below.
            if (wave == 0)
                tdm_load_2d_bf16((unsigned int)(uintptr_t)nbuf,
                                 A + (size_t)mBlk * Kd + (size_t)(s + 1) * TK,
                                 TK, TM, (unsigned int)Kd,
                                 (unsigned int)Mrows, (unsigned int)Kd);
            issue_stage_b(Bt, Kd, nBlk, (s + 1) * TK, nbuf + A_BYTES, tid);
        }

        unsigned char* sA = smem + (s & 1) * STAGE_BYTES;
        unsigned char* sB = sA + A_BYTES;

#pragma unroll
        for (int ks = 0; ks < 2; ++ks) {       // two 32-deep WMMA steps
            FragAB fa[2], fb[2];
#pragma unroll
            for (int mi = 0; mi < 2; ++mi) {
                const unsigned char* p =
                    sA + (wm * 32 + mi * 16 + ln) * LROW + ks * 64 + 16 * lg;
                fa[mi].q[0] = *(const uint4*)(p);
                fa[mi].q[1] = *(const uint4*)(p + 32);
            }
#pragma unroll
            for (int ni = 0; ni < 2; ++ni) {
                const unsigned char* p =
                    sB + (wn * 32 + ni * 16 + ln) * LROW + ks * 64 + 16 * lg;
                fb[ni].q[0] = *(const uint4*)(p);
                fb[ni].q[1] = *(const uint4*)(p + 32);
            }
#pragma unroll
            for (int mi = 0; mi < 2; ++mi)
#pragma unroll
                for (int ni = 0; ni < 2; ++ni)
                    acc[mi][ni] = __builtin_amdgcn_wmma_f32_16x16x32_bf16(
                        false, fa[mi].v, false, fb[ni].v,
                        (short)0, acc[mi][ni], false, false);
        }
        __syncthreads();        // all waves done reading buffer s&1
    }

    // epilogue: C element j -> row +j+8*lg, col = 16-tile base + ln
#pragma unroll
    for (int mi = 0; mi < 2; ++mi) {
#pragma unroll
        for (int ni = 0; ni < 2; ++ni) {
            const int col = nBlk + wn * 32 + ni * 16 + ln;
            const float bv = bias[col];
#pragma unroll
            for (int j = 0; j < 8; ++j) {
                const int row = mBlk + wm * 32 + mi * 16 + j + 8 * lg;
                float v = acc[mi][ni][j] + bv;
                if (RELU) v = v > 0.0f ? v : 0.0f;
                if (OUT_BF16)
                    ((unsigned short*)Cout)[(size_t)row * Ncols + col] = f32_to_bf16(v);
                else
                    ((float*)Cout)[(size_t)row * Ncols + col] = v;
            }
        }
    }
}

// ---------------- sequential mode scan ----------------
__global__ __launch_bounds__(32)
void k_scan(const float* __restrict__ logits,      // [B][S][Kmod]
            const float* __restrict__ prev_mode,   // [B][Kmod]
            int* __restrict__ mode_idx,            // [B*S]
            float* __restrict__ modes_out) {       // [B][S][Kmod]
    const int b = blockIdx.x;
    const int t = threadIdx.x;      // 0..31

    float bv = NEG_BIG; int bi = 0;
    for (int j = t; j < Kmod; j += 32) {
        float v = prev_mode[b * Kmod + j];
        if (v > bv) { bv = v; bi = j; }
    }
#pragma unroll
    for (int m = 16; m >= 1; m >>= 1) {
        float ov = __shfl_xor(bv, m, 32);
        int   oi = __shfl_xor(bi, m, 32);
        if (ov > bv || (ov == bv && oi < bi)) { bv = ov; bi = oi; }
    }
    int prev = bi;

    const float* base = logits + (size_t)b * Ssz * Kmod;

    float buf[8][8];   // [pipeline phase][strided j]
#pragma unroll
    for (int p = 0; p < 8; ++p)
#pragma unroll
        for (int j = 0; j < 8; ++j)
            buf[p][j] = base[(size_t)p * Kmod + j * 32 + t];

    for (int so = 0; so < Ssz; so += 8) {
#pragma unroll
        for (int p = 0; p < 8; ++p) {
            const int s = so + p;
            float mv = NEG_BIG; int mi = 0;
#pragma unroll
            for (int j = 0; j < 8; ++j) {
                const int idx = j * 32 + t;
                float v = buf[p][j] + (idx == prev ? STICKINESS : 0.0f);
                if (v > mv) { mv = v; mi = idx; }
            }
#pragma unroll
            for (int m = 16; m >= 1; m >>= 1) {
                float ov = __shfl_xor(mv, m, 32);
                int   oi = __shfl_xor(mi, m, 32);
                if (ov > mv || (ov == mv && oi < mi)) { mv = ov; mi = oi; }
            }
            float* mo = modes_out + ((size_t)b * Ssz + s) * Kmod;
#pragma unroll
            for (int j = 0; j < 8; ++j) {
                const int idx = j * 32 + t;
                mo[idx] = (idx == mi) ? 1.0f : 0.0f;
            }
            if (t == 0) mode_idx[b * Ssz + s] = mi;
            prev = mi;
            const int sn = s + 8;
            if (sn < Ssz) {
#pragma unroll
                for (int j = 0; j < 8; ++j)
                    buf[p][j] = base[(size_t)sn * Kmod + j * 32 + t];
            }
        }
    }
}

// ---------------- per-mode output table ----------------
__global__ __launch_bounds__(256)
void k_modes_table(const float* __restrict__ Mmat,       // [Kmod][DM]
                   const float* __restrict__ g,          // [DM]
                   const float* __restrict__ Wrd_w,      // [DM][DOUT]
                   const float* __restrict__ Wrd_b,      // [DOUT]
                   const float* __restrict__ bank_keys,  // [BANK][DM]
                   const float* __restrict__ bank_vals,  // [BANK][DM]
                   const float* __restrict__ bank_used,  // [BANK]
                   float* __restrict__ Ymodes) {         // [Kmod][DOUT]
    __shared__ float s_mem[DM];
    __shared__ float s_qn[DM];     // reused later for rn
    __shared__ float s_sc[BANK];
    __shared__ float s_red[256];

    const int k = blockIdx.x;
    const int t = threadIdx.x;

    float ss = 0.0f;
    for (int d = t; d < DM; d += 256) {
        float v = Mmat[(size_t)k * DM + d];
        s_mem[d] = v;
        ss += v * v;
    }
    s_red[t] = ss; __syncthreads();
    for (int off = 128; off > 0; off >>= 1) {
        if (t < off) s_red[t] += s_red[t + off];
        __syncthreads();
    }
    const float qinv = 1.0f / fmaxf(sqrtf(s_red[0]), 1e-12f);
    __syncthreads();
    for (int d = t; d < DM; d += 256) s_qn[d] = s_mem[d] * qinv;
    __syncthreads();

    for (int jj = 0; jj < BANK / 256; ++jj) {
        const int n = jj * 256 + t;
        const float* kp = bank_keys + (size_t)n * DM;
        float dot = 0.0f, ks = 0.0f;
        for (int d = 0; d < DM; ++d) {
            float kv = kp[d];
            dot += s_qn[d] * kv;
            ks  += kv * kv;
        }
        float sc = BANK_SCALE * dot / fmaxf(sqrtf(ks), 1e-12f);
        s_sc[n] = (bank_used[n] > 0.5f) ? sc : -1e30f;
    }
    __syncthreads();

    float mx = NEG_BIG;
    for (int jj = 0; jj < BANK / 256; ++jj) mx = fmaxf(mx, s_sc[jj * 256 + t]);
    s_red[t] = mx; __syncthreads();
    for (int off = 128; off > 0; off >>= 1) {
        if (t < off) s_red[t] = fmaxf(s_red[t], s_red[t + off]);
        __syncthreads();
    }
    mx = s_red[0]; __syncthreads();
    float sme = 0.0f;
    for (int jj = 0; jj < BANK / 256; ++jj) {
        const int n = jj * 256 + t;
        float e = __expf(s_sc[n] - mx);
        s_sc[n] = e;
        sme += e;
    }
    s_red[t] = sme; __syncthreads();
    for (int off = 128; off > 0; off >>= 1) {
        if (t < off) s_red[t] += s_red[t + off];
        __syncthreads();
    }
    const float inv = 1.0f / s_red[0];
    __syncthreads();

    float b0 = 0.0f, b1 = 0.0f;
    for (int n = 0; n < BANK; ++n) {
        const float a = s_sc[n] * inv;
        const float* vp = bank_vals + (size_t)n * DM;
        b0 += a * vp[t];
        b1 += a * vp[t + 256];
    }

    const float r0 = s_mem[t] + b0;
    const float r1 = s_mem[t + 256] + b1;
    s_red[t] = r0 * r0 + r1 * r1; __syncthreads();
    for (int off = 128; off > 0; off >>= 1) {
        if (t < off) s_red[t] += s_red[t + off];
        __syncthreads();
    }
    const float rscale = 1.0f / sqrtf(s_red[0] / (float)DM + RMS_EPS);
    __syncthreads();
    s_qn[t]       = r0 * g[t]       * rscale;
    s_qn[t + 256] = r1 * g[t + 256] * rscale;
    __syncthreads();

    for (int jj = 0; jj < DOUT / 256; ++jj) {
        const int o = jj * 256 + t;
        float accv = Wrd_b[o];
        for (int d = 0; d < DM; ++d)
            accv += s_qn[d] * Wrd_w[(size_t)d * DOUT + o];
        Ymodes[(size_t)k * DOUT + o] = accv;
    }
}

// ---------------- gather final output ----------------
__global__ __launch_bounds__(256)
void k_gather(const float* __restrict__ Ymodes,
              const int* __restrict__ mode_idx,
              float* __restrict__ y) {
    const int row = blockIdx.x;
    const int t   = threadIdx.x;
    const int idx = mode_idx[row];
    const float4* src = (const float4*)(Ymodes + (size_t)idx * DOUT);
    float4*       dst = (float4*)(y + (size_t)row * DOUT);
    dst[t] = src[t];
}

// ---------------- launch ----------------
extern "C" void kernel_launch(void* const* d_in, const int* in_sizes, int n_in,
                              void* d_out, int out_size, void* d_ws, size_t ws_size,
                              hipStream_t stream) {
    const float* x          = (const float*)d_in[0];
    const float* prev_mode  = (const float*)d_in[1];
    const float* Wtr_w      = (const float*)d_in[2];
    const float* Wtr_b      = (const float*)d_in[3];
    const float* Wms_w      = (const float*)d_in[4];
    const float* Wms_b      = (const float*)d_in[5];
    const float* Mmat       = (const float*)d_in[6];
    const float* g          = (const float*)d_in[7];
    const float* Wrd_w      = (const float*)d_in[8];
    const float* Wrd_b      = (const float*)d_in[9];
    const float* bank_keys  = (const float*)d_in[10];
    const float* bank_vals  = (const float*)d_in[11];
    const float* bank_used  = (const float*)d_in[12];

    float* y_out     = (float*)d_out;                       // [MTOT][DOUT]
    float* modes_out = y_out + (size_t)MTOT * DOUT;         // [MTOT][Kmod]

    char* ws = (char*)d_ws;
    size_t off = 0;
    unsigned short* xb   = (unsigned short*)(ws + off); off += (size_t)MTOT * DIN  * 2;
    unsigned short* WtrT = (unsigned short*)(ws + off); off += (size_t)Hdim * DIN  * 2;
    unsigned short* WmsT = (unsigned short*)(ws + off); off += (size_t)Kmod * Hdim * 2;
    unsigned short* hbuf = (unsigned short*)(ws + off); off += (size_t)MTOT * Hdim * 2;
    float* logits        = (float*)(ws + off);          off += (size_t)MTOT * Kmod * 4;
    int*   mode_idx      = (int*)(ws + off);            off += (size_t)MTOT * 4;
    float* Ymodes        = (float*)(ws + off);          off += (size_t)Kmod * DOUT * 4;

    // 1) precision conversion / weight transposes
    k_f32_to_bf16 <<<4096, 256, 0, stream>>>(x, xb, MTOT * DIN);
    k_transpose_bf16<<<2048, 256, 0, stream>>>(Wtr_w, WtrT, DIN, Hdim);
    k_transpose_bf16<<<512,  256, 0, stream>>>(Wms_w, WmsT, Hdim, Kmod);

    // 2) h = relu(x @ Wtr + b)  -> bf16
    k_gemm_bf16<true, true><<<dim3(MTOT / TM, Hdim / TN), 256, 0, stream>>>(
        xb, WtrT, Wtr_b, hbuf, MTOT, Hdim, DIN);

    // 3) logits = h @ Wms + b  -> f32
    k_gemm_bf16<false, false><<<dim3(MTOT / TM, Kmod / TN), 256, 0, stream>>>(
        hbuf, WmsT, Wms_b, logits, MTOT, Kmod, Hdim);

    // 4) sequential sticky-argmax scan (writes one-hot modes to d_out)
    k_scan<<<Bsz, 32, 0, stream>>>(logits, prev_mode, mode_idx, modes_out);

    // 5) per-mode bank attention + RMSNorm + readout table
    k_modes_table<<<Kmod, 256, 0, stream>>>(Mmat, g, Wrd_w, Wrd_b,
                                            bank_keys, bank_vals, bank_used, Ymodes);

    // 6) y[row] = Ymodes[mode_idx[row]]
    k_gather<<<MTOT, 256, 0, stream>>>(Ymodes, mode_idx, y_out);
}
